// TransferNet_89395449299191
// MI455X (gfx1250) — matrix operand order repaired and checked by reading, not compiled, not run on hardware
//
#include <hip/hip_runtime.h>

// ---------------------------------------------------------------------------
// TransferNet on MI455X (gfx1250): fused bf16-WMMA BiGRU + streaming tail.
// ---------------------------------------------------------------------------

typedef __attribute__((ext_vector_type(16))) __bf16 v16bf;
typedef __attribute__((ext_vector_type(8)))  __bf16 v8bf;
typedef __attribute__((ext_vector_type(4)))  __bf16 v4bf;
typedef __attribute__((ext_vector_type(8)))  float  v8f;
typedef __attribute__((ext_vector_type(4)))  float  v4f;
typedef __attribute__((ext_vector_type(4)))  int    v4i;

static constexpr int Bq = 8, Sq = 32, Ld = 16, Aq = 400, Tq = 2;
static constexpr int DW = 256, DH = 256, Hh = 128, G3 = 384;
static constexpr int NW = 50000, NE = 200000;
static constexpr int NT = G3 / 16;   // 24 N-tiles of 16

__device__ __forceinline__ float sigf(float x) { return 1.f / (1.f + __expf(-x)); }

// ---- CDNA5 async memory->LDS path (probe-guarded; falls back gracefully) ----
#if __has_builtin(__builtin_amdgcn_global_load_async_to_lds_b128)
#define HAS_ASYNC_LDS 1
#else
#define HAS_ASYNC_LDS 0
#endif

#if HAS_ASYNC_LDS
typedef __attribute__((address_space(1))) v4i as1_v4i;
typedef __attribute__((address_space(3))) v4i as3_v4i;
__device__ __forceinline__ void async_ld16(const void* g, void* l) {
    __builtin_amdgcn_global_load_async_to_lds_b128((as1_v4i*)g, (as3_v4i*)l, 0, 0);
}
__device__ __forceinline__ void async_wait0() {
#if __has_builtin(__builtin_amdgcn_s_wait_asynccnt)
    __builtin_amdgcn_s_wait_asynccnt(0);
#else
    asm volatile("s_wait_asynccnt 0x0" ::: "memory");
#endif
}
#endif

// ---------------------------------------------------------------------------
// fp32 -> bf16 embedding conversion (vectorized x4)
// ---------------------------------------------------------------------------
__global__ void k_conv_bf16(const float* __restrict__ src, __bf16* __restrict__ dst, int n4) {
    int i = blockIdx.x * 256 + threadIdx.x;
    if (i >= n4) return;
    v4f f = ((const v4f*)src)[i];
    v4bf o;
    o[0] = (__bf16)f.x; o[1] = (__bf16)f.y; o[2] = (__bf16)f.z; o[3] = (__bf16)f.w;
    ((v4bf*)dst)[i] = o;
}

// ---------------------------------------------------------------------------
// Swizzle a [K x 384] fp32 weight matrix into WMMA bf16 B-fragment order:
// per (kt,nt) 16x16 tile-pair block of 1024B, lane l holds 16 contiguous bf16:
//   n = nt*16 + (l&15),  k = kt*32 + (l>>4)*16 + j   (j = 0..15)
// ---------------------------------------------------------------------------
__global__ void k_swizzleW(const float* __restrict__ src, __bf16* __restrict__ dst, int K) {
    int idx = blockIdx.x * 256 + threadIdx.x;
    if (idx >= K * G3) return;
    int block = idx >> 9;          // 512 bf16 per (kt,nt) block
    int within = idx & 511;
    int lane = within >> 4;
    int j = within & 15;
    int kt = block / NT;
    int nt = block - kt * NT;
    int k = kt * 32 + (lane >> 4) * 16 + j;
    int n = nt * 16 + (lane & 15);
    dst[idx] = (__bf16)src[k * G3 + n];
}

// ---------------------------------------------------------------------------
// id gathers (direction-flipped word-id streams)
// ---------------------------------------------------------------------------
__global__ void k_qids(const int* __restrict__ questions, int* __restrict__ fwd, int* __restrict__ bwd) {
    int idx = blockIdx.x * 256 + threadIdx.x;
    if (idx >= Bq * Sq) return;
    int b = idx / Sq, s = idx - b * Sq;
    int id = questions[idx];
    fwd[idx] = id;
    bwd[b * Sq + (Sq - 1 - s)] = id;
}

__global__ void k_descids(const int* __restrict__ rg, const int* __restrict__ kb_desc,
                          int t, int* __restrict__ fwd, int* __restrict__ bwd) {
    int idx = blockIdx.x * 256 + threadIdx.x;
    if (idx >= Bq * Aq * Ld) return;
    int seq = idx / Ld, l = idx - seq * Ld;
    int row = rg[t * Bq * Aq + seq];
    int id = kb_desc[(size_t)row * Ld + l];
    fwd[idx] = id;
    bwd[seq * Ld + (Ld - 1 - l)] = id;
}

// ---------------------------------------------------------------------------
// A-fragment (16-bit 16x32, contiguous k = kt*32 + khalf*16 + s, as in B swizzle): lane m = l&15, khalf = l>>4.
// ---------------------------------------------------------------------------
__device__ __forceinline__ v16bf load_afrag(const __bf16* base, int row_stride_bytes,
                                            int m, int kt, int khalf) {
    const char* p = (const char*)base + m * row_stride_bytes + kt * 64 + khalf * 32;
    v8bf c0 = *(const v8bf*)p;
    v8bf c1 = *(const v8bf*)(p + 16);
    return __builtin_shufflevector(c0, c1, 0,1,2,3,4,5,6,7,8,9,10,11,12,13,14,15);
}

// ---------------------------------------------------------------------------
// Fused GRU: 16 sequences per workgroup, 8 waves, wave w owns hidden cols
// [16w,16w+16) of r/z/n (n-tiles w, w+8, w+16).  h kept fp32 in registers,
// bf16 shadow in LDS feeds h@Whh WMMA.  Weight fragments are loop-invariant
// -> the compiler keeps all 36 B-fragments register-resident (VGPR-MSB).
// x_t tiles staged memory->LDS via CDNA5 async loads when available.
// ---------------------------------------------------------------------------
__global__ void __launch_bounds__(256)
k_gru(const int* __restrict__ ids, const __bf16* __restrict__ emb,
      const __bf16* __restrict__ Wih, const __bf16* __restrict__ Whh,
      const float* __restrict__ bias, float* __restrict__ h_all,
      float* __restrict__ h_last, int nseq, int Lw, int store_all) {
    __shared__ __align__(16) __bf16 s_x[16 * 256];   // 8 KB: 16 rows of x_t (bf16)
    __shared__ __align__(16) __bf16 s_h[16 * 128];   // 4 KB: 16 rows of h (bf16 shadow)

    const int tid = threadIdx.x;
    const int wv = tid >> 5;
    const int lane = tid & 31;
    const int khalf = lane >> 4;
    const int nn = lane & 15;
    const int col = wv * 16 + nn;                    // hidden column 0..127
    const int seq0 = blockIdx.x * 16;

    const float br = bias[col];
    const float bz = bias[128 + col];
    const float bn = bias[256 + col];

    const v16bf* WihV = (const v16bf*)Wih;           // 32 v16bf per (kt,nt) block
    const v16bf* WhhV = (const v16bf*)Whh;

    float hreg[8];
#pragma unroll
    for (int i = 0; i < 8; ++i) hreg[i] = 0.f;
    for (int i = tid; i < 16 * 128; i += 256) s_h[i] = (__bf16)0.f;
    __syncthreads();

    for (int l = 0; l < Lw; ++l) {
        // ---- stage x_t rows: gather 16 word rows (512B each) into LDS ----
#if HAS_ASYNC_LDS
        {
#pragma unroll
            for (int it = 0; it < 2; ++it) {
                int c = tid + it * 256;
                int m = c >> 5;               // 32 chunks of 16B per 512B row
                int co = c & 31;
                int sq = seq0 + m;
                int id = (sq < nseq) ? ids[sq * Lw + l] : 0;
                const char* g = (const char*)(emb + (size_t)id * 256) + co * 16;
                async_ld16(g, (char*)s_x + c * 16);
            }
            async_wait0();
        }
#else
        {
            uint4* dst = (uint4*)s_x;
            for (int c = tid; c < 512; c += 256) {
                int m = c >> 5;
                int co = c & 31;
                int sq = seq0 + m;
                int id = (sq < nseq) ? ids[sq * Lw + l] : 0;
                const uint4* srow = (const uint4*)(emb + (size_t)id * 256);
                dst[c] = srow[co];
            }
        }
#endif
        __syncthreads();

        // ---- gi = x_t @ Wih (K=256, 8 k-steps) ----
        v8f gi0 = {0,0,0,0,0,0,0,0}, gi1 = {0,0,0,0,0,0,0,0}, gi2 = {0,0,0,0,0,0,0,0};
#pragma unroll
        for (int kt = 0; kt < 8; ++kt) {
            v16bf a = load_afrag(s_x, 512, nn, kt, khalf);
            v16bf b0 = WihV[(size_t)(kt * NT + wv) * 32 + lane];
            v16bf b1 = WihV[(size_t)(kt * NT + wv + 8) * 32 + lane];
            v16bf b2 = WihV[(size_t)(kt * NT + wv + 16) * 32 + lane];
            gi0 = __builtin_amdgcn_wmma_f32_16x16x32_bf16(false, a, false, b0, (short)0, gi0, false, false);
            gi1 = __builtin_amdgcn_wmma_f32_16x16x32_bf16(false, a, false, b1, (short)0, gi1, false, false);
            gi2 = __builtin_amdgcn_wmma_f32_16x16x32_bf16(false, a, false, b2, (short)0, gi2, false, false);
        }

        // ---- gh = h @ Whh (K=128, 4 k-steps) ----
        v8f gh0 = {0,0,0,0,0,0,0,0}, gh1 = {0,0,0,0,0,0,0,0}, gh2 = {0,0,0,0,0,0,0,0};
#pragma unroll
        for (int kt = 0; kt < 4; ++kt) {
            v16bf a = load_afrag(s_h, 256, nn, kt, khalf);
            v16bf b0 = WhhV[(size_t)(kt * NT + wv) * 32 + lane];
            v16bf b1 = WhhV[(size_t)(kt * NT + wv + 8) * 32 + lane];
            v16bf b2 = WhhV[(size_t)(kt * NT + wv + 16) * 32 + lane];
            gh0 = __builtin_amdgcn_wmma_f32_16x16x32_bf16(false, a, false, b0, (short)0, gh0, false, false);
            gh1 = __builtin_amdgcn_wmma_f32_16x16x32_bf16(false, a, false, b1, (short)0, gh1, false, false);
            gh2 = __builtin_amdgcn_wmma_f32_16x16x32_bf16(false, a, false, b2, (short)0, gh2, false, false);
        }
        __syncthreads();   // all s_h / s_x reads done before overwrite

        // ---- gate math (register-local for this wave's 16 columns) ----
#pragma unroll
        for (int i = 0; i < 8; ++i) {
            float r = sigf(gi0[i] + br + gh0[i]);
            float z = sigf(gi1[i] + bz + gh1[i]);
            float n = tanhf(gi2[i] + bn + r * gh2[i]);
            hreg[i] = (1.f - z) * n + z * hreg[i];
        }
        // C layout: vgpr i, lane l -> m = i + 8*(l>>4), n = l&15
#pragma unroll
        for (int i = 0; i < 8; ++i) {
            int m = i + 8 * khalf;
            s_h[m * 128 + col] = (__bf16)hreg[i];
            if (store_all && (seq0 + m) < nseq)
                h_all[((size_t)(seq0 + m) * Lw + l) * Hh + col] = hreg[i];
        }
        __syncthreads();   // h shadow visible for next step
    }

#pragma unroll
    for (int i = 0; i < 8; ++i) {
        int m = i + 8 * khalf;
        if ((seq0 + m) < nseq) h_last[(size_t)(seq0 + m) * Hh + col] = hreg[i];
    }
}

// ---------------------------------------------------------------------------
// Assemble q_word_h [8,32,256] (fwd || time-reversed bwd) and q_emb [8,256]
// ---------------------------------------------------------------------------
__global__ void k_combine_q(const float* __restrict__ hsf, const float* __restrict__ hsb,
                            const float* __restrict__ hlf, const float* __restrict__ hlb,
                            float* __restrict__ qwh, float* __restrict__ qemb) {
    int idx = blockIdx.x * 256 + threadIdx.x;
    if (idx >= Bq * Sq * DH) return;
    int b = idx >> 13;
    int rem = idx & 8191;
    int s = rem >> 8;
    int d = rem & 255;
    float v = (d < Hh) ? hsf[((b * Sq + s) * Hh) + d]
                       : hsb[((b * Sq + (Sq - 1 - s)) * Hh) + (d - Hh)];
    qwh[idx] = v;
    if (idx < Bq * DH) {
        int b2 = idx >> 8, d2 = idx & 255;
        qemb[idx] = (d2 < Hh) ? hlf[b2 * Hh + d2] : hlb[b2 * Hh + d2 - Hh];
    }
}

// ---------------------------------------------------------------------------
// Per-step attention: cq = tanh(q_emb @ W + b); softmax attn over q_word_h;
// ctxw = (ctx + cq) * rel_w   (rel_w folded in for the edge kernel)
// ---------------------------------------------------------------------------
__global__ void __launch_bounds__(256)
k_attend(const float* __restrict__ qemb, const float* __restrict__ qwh,
         const float* __restrict__ W, const float* __restrict__ bias,
         const float* __restrict__ rel_w, float* __restrict__ ctxw) {
    __shared__ float s_qe[Bq * DH];
    __shared__ float s_cq[Bq * DH];
    __shared__ float s_lg[Bq * Sq];
    __shared__ float s_sum[Bq];
    int tid = threadIdx.x;
    for (int i = tid; i < Bq * DH; i += 256) s_qe[i] = qemb[i];
    __syncthreads();

    { // cq
        int d = tid;
        for (int b = 0; b < Bq; ++b) {
            float acc = bias[d];
            for (int k = 0; k < DH; ++k) acc += s_qe[b * DH + k] * W[k * DH + d];
            s_cq[b * DH + d] = tanhf(acc);
        }
    }
    __syncthreads();
    { // logits
        int b = tid >> 5, s = tid & 31;
        float acc = 0.f;
        const float* row = qwh + ((size_t)(b * Sq + s)) * DH;
        for (int d = 0; d < DH; ++d) acc += s_cq[b * DH + d] * row[d];
        s_lg[tid] = acc;
    }
    __syncthreads();
    if (tid < Bq) { // softmax per batch row
        float mx = -1e30f;
        for (int s = 0; s < Sq; ++s) mx = fmaxf(mx, s_lg[tid * Sq + s]);
        float sm = 0.f;
        for (int s = 0; s < Sq; ++s) {
            float e = __expf(s_lg[tid * Sq + s] - mx);
            s_lg[tid * Sq + s] = e;
            sm += e;
        }
        s_sum[tid] = sm;
    }
    __syncthreads();
    { // ctx + fold rel_w
        int d = tid;
        for (int b = 0; b < Bq; ++b) {
            float c = s_cq[b * DH + d];
            float inv = 1.f / s_sum[b];
            for (int s = 0; s < Sq; ++s)
                c += (s_lg[b * Sq + s] * inv) * qwh[((size_t)(b * Sq + s)) * DH + d];
            ctxw[b * DH + d] = c * rel_w[d];
        }
    }
}

// ---------------------------------------------------------------------------
// Edge kernel: p = sigmoid(<desc_emb, ctx*rel_w> + rel_b); scatter-add
// ---------------------------------------------------------------------------
__global__ void k_edge(const int* __restrict__ rg, const int* __restrict__ kb_pair,
                       const float* __restrict__ hlf, const float* __restrict__ hlb,
                       const float* __restrict__ ctxw, const float* __restrict__ rel_b,
                       const float* __restrict__ eprev, float* __restrict__ enext, int t) {
    int idx = blockIdx.x * 256 + threadIdx.x;
    if (idx >= Bq * Aq) return;
    int b = idx / Aq;
    int eg = rg[t * Bq * Aq + idx];
    int p0 = kb_pair[2 * eg];
    int p1 = kb_pair[2 * eg + 1];
    const v4f* hf = (const v4f*)(hlf + (size_t)idx * Hh);
    const v4f* hb = (const v4f*)(hlb + (size_t)idx * Hh);
    const v4f* cw = (const v4f*)(ctxw + b * DH);
    float acc = 0.f;
#pragma unroll 4
    for (int d = 0; d < Hh / 4; ++d) {
        v4f a = hf[d], c = cw[d];
        acc += a.x * c.x + a.y * c.y + a.z * c.z + a.w * c.w;
    }
#pragma unroll 4
    for (int d = 0; d < Hh / 4; ++d) {
        v4f a = hb[d], c = cw[Hh / 4 + d];
        acc += a.x * c.x + a.y * c.y + a.z * c.z + a.w * c.w;
    }
    float pr = sigf(acc + rel_b[0]);
    atomicAdd(&enext[(size_t)b * NE + p1], eprev[(size_t)b * NE + p0] * pr);
}

__global__ void k_zero4(v4f* __restrict__ p, int n4) {
    int i = blockIdx.x * 256 + threadIdx.x;
    if (i < n4) p[i] = (v4f){0.f, 0.f, 0.f, 0.f};
}

__global__ void k_norm4(v4f* __restrict__ p, int n4) {
    int i = blockIdx.x * 256 + threadIdx.x;
    if (i < n4) {
        v4f v = p[i];
        v.x = v.x > 1.f ? 1.f : v.x;
        v.y = v.y > 1.f ? 1.f : v.y;
        v.z = v.z > 1.f ? 1.f : v.z;
        v.w = v.w > 1.f ? 1.f : v.w;
        p[i] = v;
    }
}

// ---------------------------------------------------------------------------
// Tail: out[b,e] = last_e[b,e] * sigmoid(q_emb[b,:] . q_w[:,e] + q_b[e])
// q_w streamed (205MB) float4/thread with NT loads + prefetch; q_emb reads
// are lane-uniform (scalar path).
// ---------------------------------------------------------------------------
__global__ void __launch_bounds__(256)
k_final(const float* __restrict__ laste, const float* __restrict__ qemb,
        const float* __restrict__ q_w, const float* __restrict__ q_b,
        float* __restrict__ out) {
    int v = blockIdx.x * 256 + threadIdx.x;     // one float4 column group
    if (v >= NE / 4) return;
    int e = v * 4;
    v4f acc[Bq];
#pragma unroll
    for (int b = 0; b < Bq; ++b) acc[b] = (v4f){0.f, 0.f, 0.f, 0.f};
    for (int k = 0; k < DH; ++k) {
        if (k + 8 < DH) __builtin_prefetch(&q_w[(size_t)(k + 8) * NE + e], 0, 0);
        const v4f* wp = (const v4f*)(q_w + (size_t)k * NE) + v;
        v4f w = __builtin_nontemporal_load(wp);
#pragma unroll
        for (int b = 0; b < Bq; ++b) {
            float qe = qemb[b * DH + k];        // lane-uniform -> s_load
            acc[b] += w * qe;
        }
    }
    v4f qb4 = ((const v4f*)q_b)[v];
#pragma unroll
    for (int b = 0; b < Bq; ++b) {
        v4f le = ((const v4f*)(laste + (size_t)b * NE))[v];
        v4f r;
        r.x = le.x * sigf(acc[b].x + qb4.x);
        r.y = le.y * sigf(acc[b].y + qb4.y);
        r.z = le.z * sigf(acc[b].z + qb4.z);
        r.w = le.w * sigf(acc[b].w + qb4.w);
        ((v4f*)(out + (size_t)b * NE))[v] = r;
    }
}

// ---------------------------------------------------------------------------
extern "C" void kernel_launch(void* const* d_in, const int* in_sizes, int n_in,
                              void* d_out, int out_size, void* d_ws, size_t ws_size,
                              hipStream_t stream) {
    (void)in_sizes; (void)n_in; (void)out_size; (void)ws_size;
    const int*   questions = (const int*)d_in[0];
    const float* e_s       = (const float*)d_in[1];
    const int*   rg        = (const int*)d_in[2];
    const int*   kb_pair   = (const int*)d_in[3];
    const int*   kb_desc   = (const int*)d_in[4];
    const float* word_emb  = (const float*)d_in[5];
    const float* qWih      = (const float*)d_in[6];
    const float* qWhh      = (const float*)d_in[7];
    const float* qb        = (const float*)d_in[8];
    const float* dWih      = (const float*)d_in[9];
    const float* dWhh      = (const float*)d_in[10];
    const float* db        = (const float*)d_in[11];
    const float* step_W    = (const float*)d_in[12];
    const float* step_b    = (const float*)d_in[13];
    const float* rel_w     = (const float*)d_in[14];
    const float* rel_b     = (const float*)d_in[15];
    const float* q_w       = (const float*)d_in[16];
    const float* q_b       = (const float*)d_in[17];
    float* out = (float*)d_out;

    char* p = (char*)d_ws;
    auto carve = [&](size_t bytes) -> char* {
        char* r = p;
        p += (bytes + 255) & ~(size_t)255;
        return r;
    };
    __bf16* embb   = (__bf16*)carve((size_t)NW * DW * 2);
    __bf16* qWihF  = (__bf16*)carve((size_t)2 * DW * G3 * 2);
    __bf16* qWhhF  = (__bf16*)carve((size_t)2 * Hh * G3 * 2);
    __bf16* dWihF  = (__bf16*)carve((size_t)2 * DW * G3 * 2);
    __bf16* dWhhF  = (__bf16*)carve((size_t)2 * Hh * G3 * 2);
    int* q_ids_f   = (int*)carve((size_t)Bq * Sq * 4);
    int* q_ids_b   = (int*)carve((size_t)Bq * Sq * 4);
    int* d_ids_f   = (int*)carve((size_t)Bq * Aq * Ld * 4);
    int* d_ids_b   = (int*)carve((size_t)Bq * Aq * Ld * 4);
    float* q_hs_f  = (float*)carve((size_t)Bq * Sq * Hh * 4);
    float* q_hs_b  = (float*)carve((size_t)Bq * Sq * Hh * 4);
    float* q_hl_f  = (float*)carve((size_t)Bq * Hh * 4);
    float* q_hl_b  = (float*)carve((size_t)Bq * Hh * 4);
    float* d_hl_f  = (float*)carve((size_t)Bq * Aq * Hh * 4);
    float* d_hl_b  = (float*)carve((size_t)Bq * Aq * Hh * 4);
    float* q_wh    = (float*)carve((size_t)Bq * Sq * DH * 4);
    float* q_emb   = (float*)carve((size_t)Bq * DH * 4);
    float* ctxw    = (float*)carve((size_t)Bq * DH * 4);
    float* ebufA   = (float*)carve((size_t)Bq * NE * 4);
    float* ebufB   = (float*)carve((size_t)Bq * NE * 4);

    // 1) embeddings to bf16
    k_conv_bf16<<<(NW * DW / 4 + 255) / 256, 256, 0, stream>>>(word_emb, embb, NW * DW / 4);

    // 2) weight swizzles (bf16 fragment order)
    struct WS { const float* s; __bf16* d; int K; };
    WS wsl[8] = {
        {qWih,               qWihF,            DW}, {qWih + DW * G3, qWihF + DW * G3, DW},
        {qWhh,               qWhhF,            Hh}, {qWhh + Hh * G3, qWhhF + Hh * G3, Hh},
        {dWih,               dWihF,            DW}, {dWih + DW * G3, dWihF + DW * G3, DW},
        {dWhh,               dWhhF,            Hh}, {dWhh + Hh * G3, dWhhF + Hh * G3, Hh},
    };
    for (int i = 0; i < 8; ++i)
        k_swizzleW<<<(wsl[i].K * G3 + 255) / 256, 256, 0, stream>>>(wsl[i].s, wsl[i].d, wsl[i].K);

    // 3) question encoding
    k_qids<<<1, 256, 0, stream>>>(questions, q_ids_f, q_ids_b);
    k_gru<<<1, 256, 0, stream>>>(q_ids_f, embb, qWihF,           qWhhF,           qb,      q_hs_f, q_hl_f, Bq, Sq, 1);
    k_gru<<<1, 256, 0, stream>>>(q_ids_b, embb, qWihF + DW * G3, qWhhF + Hh * G3, qb + G3, q_hs_b, q_hl_b, Bq, Sq, 1);
    k_combine_q<<<(Bq * Sq * DH + 255) / 256, 256, 0, stream>>>(q_hs_f, q_hs_b, q_hl_f, q_hl_b, q_wh, q_emb);

    // 4) transfer steps
    for (int t = 0; t < Tq; ++t) {
        k_attend<<<1, 256, 0, stream>>>(q_emb, q_wh, step_W + (size_t)t * DH * DH, step_b + (size_t)t * DH,
                                        rel_w, ctxw);
        k_descids<<<(Bq * Aq * Ld + 255) / 256, 256, 0, stream>>>(rg, kb_desc, t, d_ids_f, d_ids_b);
        k_gru<<<(Bq * Aq) / 16, 256, 0, stream>>>(d_ids_f, embb, dWihF,           dWhhF,           db,      d_hl_f, d_hl_f, Bq * Aq, Ld, 0);
        k_gru<<<(Bq * Aq) / 16, 256, 0, stream>>>(d_ids_b, embb, dWihF + DW * G3, dWhhF + Hh * G3, db + G3, d_hl_b, d_hl_b, Bq * Aq, Ld, 0);

        const float* eprev = (t == 0) ? e_s : ebufA;
        float* enext = (t == 0) ? ebufA : ebufB;
        k_zero4<<<(Bq * NE / 4 + 255) / 256, 256, 0, stream>>>((v4f*)enext, Bq * NE / 4);
        k_edge<<<(Bq * Aq + 255) / 256, 256, 0, stream>>>(rg, kb_pair, d_hl_f, d_hl_b, ctxw, rel_b,
                                                          eprev, enext, t);
        k_norm4<<<(Bq * NE / 4 + 255) / 256, 256, 0, stream>>>((v4f*)enext, Bq * NE / 4);
    }

    // 5) final gating GEMM (memory-bound q_w stream)
    k_final<<<(NE / 4 + 255) / 256, 256, 0, stream>>>(ebufB, q_emb, q_w, q_b, out);
}
